// WFMultiHeadAttention_84713934946754
// MI455X (gfx1250) — matrix operand order repaired and checked by reading, not compiled, not run on hardware
//
#include <hip/hip_runtime.h>
#include <hip/hip_bf16.h>
#include <cstdint>

// ---------------------------------------------------------------------------
// WF Multi-Head Attention for MI455X (gfx1250, wave32, WMMA)
// B=16, S=1024, E=512, H=8, D=64, L=4.  M = B*S = 16384.
// All matmuls use v_wmma_f32_16x16x32_f16 (f16 operands, f32 accumulate).
// Q is pre-scaled by 1/D = 1/64 (covers both D^-0.5 scalings in the ref).
// Pure-copy LDS staging uses CDNA5 async global->LDS (ASYNCcnt) path.
// Workspace layout (needs 64 MiB): Qb | Kb | Vb | Ctx, 16 MiB f16 each.
// ---------------------------------------------------------------------------

typedef __attribute__((ext_vector_type(16))) _Float16 v16h;
typedef __attribute__((ext_vector_type(8)))  _Float16 v8h;
typedef __attribute__((ext_vector_type(4)))  _Float16 v4h;
typedef __attribute__((ext_vector_type(8)))  float    v8f;
typedef __attribute__((ext_vector_type(4)))  float    v4f;

#define GE 512     // E (both N and K of the projection GEMMs)
#define GS 1024    // S
#define GH 8       // heads
#define GD 64      // head dim
#define GB 16      // batch

__device__ __forceinline__ v16h ld16(const _Float16* p0, const _Float16* p1) {
  v8h lo = *(const v8h*)p0;
  v8h hi = *(const v8h*)p1;
  return __builtin_shufflevector(lo, hi, 0,1,2,3,4,5,6,7,8,9,10,11,12,13,14,15);
}

__device__ __forceinline__ v8f wmma32(v16h a, v16h b, v8f c) {
  return __builtin_amdgcn_wmma_f32_16x16x32_f16(false, a, false, b, (short)0, c,
                                                false, false);
}

// Async global->LDS copy of 16 bytes per lane (GLOBAL_LOAD_ASYNC_TO_LDS_B128,
// tracked by ASYNCcnt). ldsoff = wave-relative LDS byte offset (low 32 bits of
// the generic shared-space pointer).
__device__ __forceinline__ void async_copy_b128(unsigned ldsoff,
                                                const void* gptr) {
  asm volatile("global_load_async_to_lds_b128 %0, %1, off"
               :: "v"(ldsoff), "v"(gptr)
               : "memory");
}

__device__ __forceinline__ void async_wait0() {
  asm volatile("s_wait_asynccnt 0" ::: "memory");
}

// ---------------------------------------------------------------------------
// Generic projection GEMM:  Out[m,n] = (sum_k A[m,k] * W[n,k] + bias[n]) * scale
//   AH    : A is f16 (async-copied to LDS); else f32 (converted while staging)
//   WF    : W = Wsh * Wsp[lang], bias = bsh + bsp[lang]
//   BHSD  : store f16 into (B,H,S,D) layout (else f32 flat (M,E))
// Block tile: 128(M) x 64(N) x 32(K). 256 threads = 8 waves in a 4x2 grid,
// each wave computes a 32x32 tile as 2x2 WMMA 16x16 tiles.
// ---------------------------------------------------------------------------
template<bool AH, bool WF, bool BHSD>
__global__ __launch_bounds__(256)
void proj_gemm(const void* __restrict__ Ain,
               const float* __restrict__ Wsh,
               const float* __restrict__ Wsp,   // (L,E,E) or unused
               const float* __restrict__ bsh,
               const float* __restrict__ bsp,   // (L,E) or unused
               const int*  __restrict__ langp,
               void* __restrict__ Out,
               float scale)
{
  __shared__ alignas(16) _Float16 Xs[128 * 40];  // 128 x 32, row stride 40
  __shared__ alignas(16) _Float16 Ws[ 64 * 40];  //  64 x 32, row stride 40

  const int tid  = threadIdx.x;
  const int wave = tid >> 5, lane = tid & 31;
  const int wm   = wave >> 1, wn = wave & 1;
  const int half = lane >> 4, l16 = lane & 15;
  const int bm   = blockIdx.x * 128;
  const int bn   = blockIdx.y * 64;

  int lang = 0;
  if constexpr (WF) lang = langp[0];
  const float* WspL = WF ? (Wsp + (size_t)lang * GE * GE) : nullptr;

  v8f acc[2][2] = {};

  for (int k0 = 0; k0 < GE; k0 += 32) {
    __syncthreads();
    // ---- stage A tile (128 x 32) as f16 ----
    if constexpr (AH) {
      const _Float16* Af = (const _Float16*)Ain;
#pragma unroll
      for (int c = 0; c < 2; ++c) {
        int ch = tid * 2 + c;                 // 512 chunks of 8 halfs
        int row = ch >> 2, c8 = (ch & 3) * 8;
        unsigned lds = (unsigned)(uintptr_t)(&Xs[row * 40 + c8]);
        async_copy_b128(lds, &Af[(size_t)(bm + row) * GE + k0 + c8]);
      }
    } else {
      const float* Af = (const float*)Ain;
#pragma unroll
      for (int c = 0; c < 4; ++c) {
        int ch = tid * 4 + c;                 // 1024 chunks of float4
        int row = ch >> 3, c4 = (ch & 7) * 4;
        v4f x = *(const v4f*)&Af[(size_t)(bm + row) * GE + k0 + c4];
        v4h hx;
        hx[0] = (_Float16)x[0]; hx[1] = (_Float16)x[1];
        hx[2] = (_Float16)x[2]; hx[3] = (_Float16)x[3];
        *(v4h*)&Xs[row * 40 + c4] = hx;
      }
    }
    // ---- stage W tile (64 x 32), optionally WF-merged ----
#pragma unroll
    for (int c = 0; c < 2; ++c) {
      int ch = tid * 2 + c;                   // 512 chunks of float4
      int row = ch >> 3, c4 = (ch & 7) * 4;
      size_t off = (size_t)(bn + row) * GE + k0 + c4;
      v4f w = *(const v4f*)&Wsh[off];
      if constexpr (WF) {
        v4f s = *(const v4f*)&WspL[off];
        w = w * s;
      }
      v4h hw;
      hw[0] = (_Float16)w[0]; hw[1] = (_Float16)w[1];
      hw[2] = (_Float16)w[2]; hw[3] = (_Float16)w[3];
      *(v4h*)&Ws[row * 40 + c4] = hw;
    }
    if constexpr (AH) async_wait0();
    __syncthreads();

    if (k0 + 32 < GE) {  // hint next A tile into cache
      const char* nxt = (const char*)Ain +
          ((size_t)(bm + (tid >> 1)) * GE + k0 + 32) * (AH ? 2 : 4);
      __builtin_prefetch(nxt, 0, 0);
    }

    // ---- fragments: A interleaved-K layout, B interleaved-K layout ----
    v16h a[2], b[2];
#pragma unroll
    for (int mt = 0; mt < 2; ++mt) {
      const _Float16* base = &Xs[(wm * 32 + mt * 16 + l16) * 40];
      a[mt] = ld16(base + half * 8, base + 16 + half * 8);
    }
#pragma unroll
    for (int nt = 0; nt < 2; ++nt) {
      const _Float16* base = &Ws[(wn * 32 + nt * 16 + l16) * 40];
      b[nt] = ld16(base + half * 8, base + 16 + half * 8);
    }
#pragma unroll
    for (int mt = 0; mt < 2; ++mt)
#pragma unroll
      for (int nt = 0; nt < 2; ++nt)
        acc[mt][nt] = wmma32(a[mt], b[nt], acc[mt][nt]);
  }

  // ---- epilogue: bias, scale, layout permute ----
#pragma unroll
  for (int mt = 0; mt < 2; ++mt) {
#pragma unroll
    for (int nt = 0; nt < 2; ++nt) {
      int ng = bn + wn * 32 + nt * 16 + l16;
      float bias = bsh[ng];
      if constexpr (WF) bias += bsp[lang * GE + ng];
#pragma unroll
      for (int r = 0; r < 8; ++r) {
        int mg = bm + wm * 32 + mt * 16 + half * 8 + r;
        float v = (acc[mt][nt][r] + bias) * scale;
        if constexpr (BHSD) {
          int b_ = mg >> 10, s_ = mg & 1023;
          int h_ = ng >> 6,  d_ = ng & 63;
          ((_Float16*)Out)[(((size_t)(b_ * GH + h_)) * GS + s_) * GD + d_] =
              (_Float16)v;
        } else {
          ((float*)Out)[(size_t)mg * GE + ng] = v;
        }
      }
    }
  }
}

// ---------------------------------------------------------------------------
// Flash attention: per (b,h), 128 query rows per block, 8 waves x 16 rows.
// Q is pre-scaled by 1/D. Online softmax over key blocks of 64.
// K block is staged via async global->LDS; V is transposed manually.
// ---------------------------------------------------------------------------
__global__ __launch_bounds__(256)
void attn_kernel(const _Float16* __restrict__ Q,
                 const _Float16* __restrict__ Kb,
                 const _Float16* __restrict__ Vb,
                 const int* __restrict__ mask,       // (B,S) key mask
                 _Float16* __restrict__ ctx)          // (B,S,E) concat layout
{
  __shared__ alignas(16) _Float16 Ks[64 * 72];      // [key][d]
  __shared__ alignas(16) _Float16 Vt[64 * 72];      // [d][key] (transposed)
  __shared__ alignas(16) _Float16 Ps[8 * 16 * 72];  // per-wave P scratch

  const int tid  = threadIdx.x;
  const int wave = tid >> 5, lane = tid & 31;
  const int half = lane >> 4, l16 = lane & 15;
  const int bh = blockIdx.y;             // b*H + h
  const int b  = bh >> 3, h = bh & 7;
  const int q0 = blockIdx.x * 128;

  const _Float16* Qh = Q  + (size_t)bh * GS * GD;
  const _Float16* Kh = Kb + (size_t)bh * GS * GD;
  const _Float16* Vh = Vb + (size_t)bh * GS * GD;
  const int* mb = mask + (size_t)b * GS;

  // Q fragments for this wave's 16 rows (A layout, 2 k-steps over d)
  v16h aq[2];
  {
    const _Float16* qp = Qh + (size_t)(q0 + wave * 16 + l16) * GD;
#pragma unroll
    for (int ks = 0; ks < 2; ++ks)
      aq[ks] = ld16(qp + ks * 32 + half * 8, qp + ks * 32 + 16 + half * 8);
  }

  v8f acc[4] = {};
  float m_i[8], l_i[8];
#pragma unroll
  for (int r = 0; r < 8; ++r) { m_i[r] = -3.0e38f; l_i[r] = 0.0f; }

  _Float16* Pw = &Ps[wave * 16 * 72];

  for (int kb = 0; kb < GS; kb += 64) {
    __syncthreads();
    // ---- stage K block [64 keys][64 d] via async global->LDS ----
#pragma unroll
    for (int c = 0; c < 2; ++c) {
      int ch = tid * 2 + c;               // 512 chunks of 8 halfs
      int row = ch >> 3, c8 = (ch & 7) * 8;
      unsigned lds = (unsigned)(uintptr_t)(&Ks[row * 72 + c8]);
      async_copy_b128(lds, &Kh[(size_t)(kb + row) * GD + c8]);
    }
    // ---- stage V transposed: Vt[d][key] ----
    {
      int key = tid & 63, d0 = (tid >> 6) * 16;
      const _Float16* vp = Vh + (size_t)(kb + key) * GD + d0;
      v8h v0 = *(const v8h*)vp;
      v8h v1 = *(const v8h*)(vp + 8);
#pragma unroll
      for (int i = 0; i < 8; ++i) {
        Vt[(d0 + i)     * 72 + key] = v0[i];
        Vt[(d0 + 8 + i) * 72 + key] = v1[i];
      }
    }
    async_wait0();
    __syncthreads();

    // ---- scores S = Q K^T (per 16-key subtile), plus key mask ----
    v8f sc[4];
#pragma unroll
    for (int st = 0; st < 4; ++st) {
      v8f s = {};
#pragma unroll
      for (int ks = 0; ks < 2; ++ks) {
        const _Float16* kp = &Ks[(st * 16 + l16) * 72 + ks * 32];
        s = wmma32(aq[ks], ld16(kp + half * 8, kp + 16 + half * 8), s);
      }
      float madd = (mb[kb + st * 16 + l16] == 0) ? -1.0e9f : 0.0f;
#pragma unroll
      for (int r = 0; r < 8; ++r) s[r] += madd;
      sc[st] = s;
    }

    // ---- online softmax (row stats live per 16-lane half) ----
#pragma unroll
    for (int r = 0; r < 8; ++r) {
      float t = fmaxf(fmaxf(sc[0][r], sc[1][r]), fmaxf(sc[2][r], sc[3][r]));
#pragma unroll
      for (int xm = 8; xm >= 1; xm >>= 1) t = fmaxf(t, __shfl_xor(t, xm, 32));
      float mn = fmaxf(m_i[r], t);
      float sc_old = __expf(m_i[r] - mn);
      m_i[r] = mn;
      float rs = 0.0f;
#pragma unroll
      for (int st = 0; st < 4; ++st) {
        float p = __expf(sc[st][r] - mn);
        sc[st][r] = p;
        rs += p;
      }
#pragma unroll
      for (int xm = 8; xm >= 1; xm >>= 1) rs += __shfl_xor(rs, xm, 32);
      l_i[r] = l_i[r] * sc_old + rs;
#pragma unroll
      for (int dt = 0; dt < 4; ++dt) acc[dt][r] *= sc_old;
    }

    // ---- P (C layout) -> LDS -> A-fragment layout ----
#pragma unroll
    for (int st = 0; st < 4; ++st)
#pragma unroll
      for (int r = 0; r < 8; ++r)
        Pw[(half * 8 + r) * 72 + st * 16 + l16] = (_Float16)sc[st][r];

#pragma unroll
    for (int ks = 0; ks < 2; ++ks) {
      const _Float16* pp = &Pw[l16 * 72 + ks * 32];
      v16h ap = ld16(pp + half * 8, pp + 16 + half * 8);
#pragma unroll
      for (int dt = 0; dt < 4; ++dt) {
        const _Float16* vp = &Vt[(dt * 16 + l16) * 72 + ks * 32];
        acc[dt] = wmma32(ap, ld16(vp + half * 8, vp + 16 + half * 8), acc[dt]);
      }
    }
  }

  // ---- normalize and store ctx in (B,S,E) concat layout ----
#pragma unroll
  for (int r = 0; r < 8; ++r) {
    float inv = 1.0f / l_i[r];
    int s = q0 + wave * 16 + half * 8 + r;
    size_t rowoff = ((size_t)b * GS + s) * GE + h * GD;
#pragma unroll
    for (int dt = 0; dt < 4; ++dt)
      ctx[rowoff + dt * 16 + l16] = (_Float16)(acc[dt][r] * inv);
  }
}

// ---------------------------------------------------------------------------
extern "C" void kernel_launch(void* const* d_in, const int* in_sizes, int n_in,
                              void* d_out, int out_size, void* d_ws, size_t ws_size,
                              hipStream_t stream) {
  const float* q        = (const float*)d_in[0];
  const float* k        = (const float*)d_in[1];
  const float* v        = (const float*)d_in[2];
  const float* Wq       = (const float*)d_in[3];
  const float* bq       = (const float*)d_in[4];
  const float* Wk       = (const float*)d_in[5];
  const float* bk       = (const float*)d_in[6];
  const float* Wv_share = (const float*)d_in[7];
  const float* Wv_spec  = (const float*)d_in[8];
  const float* bv_share = (const float*)d_in[9];
  const float* bv_spec  = (const float*)d_in[10];
  const float* Wo_share = (const float*)d_in[11];
  const float* Wo_spec  = (const float*)d_in[12];
  const float* bo_share = (const float*)d_in[13];
  const float* bo_spec  = (const float*)d_in[14];
  const int*   mask     = (const int*)d_in[15];
  const int*   lang     = (const int*)d_in[16];

  // workspace: 4 x 16 MiB f16 buffers
  char* ws = (char*)d_ws;
  _Float16* Qb  = (_Float16*)(ws + 0ull * (1ull << 24));
  _Float16* Kbf = (_Float16*)(ws + 1ull * (1ull << 24));
  _Float16* Vbf = (_Float16*)(ws + 2ull * (1ull << 24));
  _Float16* Ctx = (_Float16*)(ws + 3ull * (1ull << 24));

  dim3 blk(256);
  dim3 gproj(128, 8);   // M/128 x E/64
  dim3 gattn(8, 128);   // S/128 x (B*H)

  // Q/K/V projections (Q carries the full 1/D softmax scaling)
  proj_gemm<false, false, true><<<gproj, blk, 0, stream>>>(
      (const void*)q, Wq, nullptr, bq, nullptr, lang, (void*)Qb, 1.0f / 64.0f);
  proj_gemm<false, false, true><<<gproj, blk, 0, stream>>>(
      (const void*)k, Wk, nullptr, bk, nullptr, lang, (void*)Kbf, 1.0f);
  proj_gemm<false, true, true><<<gproj, blk, 0, stream>>>(
      (const void*)v, Wv_share, Wv_spec, bv_share, bv_spec, lang, (void*)Vbf, 1.0f);

  // attention
  attn_kernel<<<gattn, blk, 0, stream>>>(Qb, Kbf, Vbf, mask, Ctx);

  // output projection (WF-merged Wo), f32 output
  proj_gemm<true, true, false><<<gproj, blk, 0, stream>>>(
      (const void*)Ctx, Wo_share, Wo_spec, bo_share, bo_spec, lang, d_out, 1.0f);
}
